// MNISTRNN_30537217474996
// MI455X (gfx1250) — matrix operand order repaired and checked by reading, not compiled, not run on hardware
//
#include <hip/hip_runtime.h>
#include <cstdint>
#include <cstddef>

// Problem sizes (match reference)
constexpr int NB = 4096;   // batch
constexpr int NT = 28;     // timesteps
constexpr int NI = 28;     // input features
constexpr int NH = 1024;   // hidden
constexpr int NG = 4096;   // 4*NH (gates)
constexpr int NO = 10;     // output classes

typedef __bf16 bf16x16 __attribute__((ext_vector_type(16)));
typedef float  f32x8   __attribute__((ext_vector_type(8)));
typedef uint32_t uint32x4 __attribute__((ext_vector_type(4)));

union FragAB { bf16x16 v; uint32_t u[8]; };

static __device__ __forceinline__ unsigned short f32_to_bf16(float f) {
  uint32_t u = __float_as_uint(f);
  u += 0x7fffu + ((u >> 16) & 1u);        // round-to-nearest-even
  return (unsigned short)(u >> 16);
}
static __device__ __forceinline__ float bf16_to_f32(unsigned short h) {
  return __uint_as_float(((uint32_t)h) << 16);
}
static __device__ __forceinline__ float sigmoid_f(float x) {
  return 1.0f / (1.0f + __expf(-x));
}
static __device__ __forceinline__ float tanh_f(float x) {
  float e = __expf(2.0f * x);
  return (e - 1.0f) / (e + 1.0f);
}

// ---------- prep kernels ----------
__global__ __launch_bounds__(256) void cvt_f32_to_bf16_kernel(
    const float* __restrict__ in, unsigned short* __restrict__ out, int n) {
  int i = blockIdx.x * 256 + threadIdx.x;
  if (i < n) out[i] = f32_to_bf16(in[i]);
}

// x[B][T][NI] fp32 -> xb[T][B][NI] bf16 (contiguous per-timestep slab)
__global__ __launch_bounds__(256) void pack_x_kernel(
    const float* __restrict__ x, unsigned short* __restrict__ xb) {
  int i = blockIdx.x * 256 + threadIdx.x;
  if (i >= NT * NB * NI) return;
  int t = i / (NB * NI);
  int rem = i % (NB * NI);
  int b = rem / NI;
  int k = rem % NI;
  xb[i] = f32_to_bf16(x[(size_t)b * NT * NI + (size_t)t * NI + k]);
}

// ---------- LSTM step: one block computes a 128x32 tile of h/c ----------
// gates = h_prev @ W_hh + x_t @ W_ih + b ; K = 1024 (+28 padded to 32)
// 8 waves: wm in 0..3 covers 32 rows each (2 m-tiles of 16), wn in 0..1 covers 16 cols.
__global__ __launch_bounds__(256) void lstm_step_kernel(
    const unsigned short* __restrict__ h_prev,   // [NB][NH] bf16
    const unsigned short* __restrict__ x_t,      // [NB][NI] bf16
    const unsigned short* __restrict__ Whh,      // [NH][NG] bf16
    const unsigned short* __restrict__ Wih,      // [NI][NG] bf16
    const float*          __restrict__ bias,     // [NG]
    float*                __restrict__ cbuf,     // [NB][NH] fp32, in/out
    unsigned short*       __restrict__ h_out)    // [NB][NH] bf16
{
  // stride 40 halfwords (80B rows): 16B-aligned for b128 LDS ops and
  // bank-conflict-free for the 16-lane fragment reads ((20*row) mod 64
  // yields disjoint 4-bank groups).
  __shared__ unsigned short sA[2][128][40];      // double-buffered A tile [m][k]
  __shared__ unsigned short sW[2][4][32][40];    // double-buffered W, [gate][n][k]

  const int tid  = threadIdx.x;
  const int lane = tid & 31;
  const int wid  = tid >> 5;       // 8 waves
  const int wm   = wid & 3;        // 32-row slab
  const int wn   = wid >> 2;       // 16-col slab (0..1)
  const int half = lane >> 4;
  const int l16  = lane & 15;

  const int rowBase = blockIdx.x * 128;
  const int colBase = blockIdx.y * 32;

  // per-thread W-loader cell: one (gate, k-octet, n-pair)
  const int wl_g  = tid >> 6;        // 0..3
  const int wl_ko = (tid >> 4) & 3;  // 0..3, k-octet (8 rows)
  const int wl_np = tid & 15;        // 0..15, n pair

  f32x8 acc[4][2];
  #pragma unroll
  for (int g = 0; g < 4; ++g)
    #pragma unroll
    for (int mi = 0; mi < 2; ++mi)
      #pragma unroll
      for (int r = 0; r < 8; ++r) acc[g][mi][r] = 0.0f;

  // --- async A-tile load: 128 rows x 32 k = 8KB, b128 per lane, 2 per thread
  auto asyncA = [&](int kt, int b) {
    #pragma unroll
    for (int j = 0; j < 2; ++j) {
      int p = tid + j * 256;                 // 0..511 chunks of 16B
      int row = p >> 2, q = p & 3;
      const uint32_t* ga =
          (const uint32_t*)(h_prev + (size_t)(rowBase + row) * NH + kt * 32 + q * 8);
      uint32_t lo = (uint32_t)(uintptr_t)(void*)&sA[b][row][q * 8];
      asm volatile("global_load_async_to_lds_b128 %0, %1, off"
                   :: "v"(lo), "v"(ga) : "memory");
    }
  };

  // --- x-chunk A tile (K=28 padded to 32): register staged, zero-filled
  auto loadAx = [&](uint32_t ra[8]) {
    #pragma unroll
    for (int j = 0; j < 8; ++j) {
      int p = tid + j * 256;                 // 0..2047 dword pairs
      int row = p >> 4, kp = p & 15;
      ra[j] = 0u;
      if (kp * 2 < NI)
        ra[j] = *(const uint32_t*)(x_t + (size_t)(rowBase + row) * NI + kp * 2);
    }
  };
  auto storeAx = [&](int b, const uint32_t ra[8]) {
    #pragma unroll
    for (int j = 0; j < 8; ++j) {
      int p = tid + j * 256;
      int row = p >> 4, kp = p & 15;
      *(uint32_t*)&sA[b][row][kp * 2] = ra[j];
    }
  };

  // --- W tiles for 4 gates: thread loads 8 K-rows of one column pair
  auto loadW = [&](int kt, uint32_t rw[8]) {
    const bool isX = (kt == 32);
    const int  k0  = kt * 32;
    const int  coff = wl_g * NH + colBase + wl_np * 2;
    #pragma unroll
    for (int i = 0; i < 8; ++i) {
      int k = wl_ko * 8 + i;
      rw[i] = 0u;
      if (!isX)
        rw[i] = *(const uint32_t*)(Whh + (size_t)(k0 + k) * NG + coff);
      else if (k < NI)
        rw[i] = *(const uint32_t*)(Wih + (size_t)k * NG + coff);
    }
  };
  // transpose in registers, two 16B stores: rows [n][k0..k0+7], [n+1][k0..k0+7]
  auto storeW = [&](int b, const uint32_t rw[8]) {
    uint32x4 lov, hiv;
    #pragma unroll
    for (int i = 0; i < 4; ++i) {
      lov[i] = (rw[2 * i] & 0xffffu) | (rw[2 * i + 1] << 16);
      hiv[i] = (rw[2 * i] >> 16) | (rw[2 * i + 1] & 0xffff0000u);
    }
    *(uint32x4*)&sW[b][wl_g][wl_np * 2 + 0][wl_ko * 8] = lov;
    *(uint32x4*)&sW[b][wl_g][wl_np * 2 + 1][wl_ko * 8] = hiv;
  };

  // --- 8 WMMAs on one buffered chunk
  auto compute = [&](int b) {
    FragAB fa[2];
    #pragma unroll
    for (int mi = 0; mi < 2; ++mi) {
      const int m = wm * 32 + mi * 16 + l16;
      #pragma unroll
      for (int v = 0; v < 8; ++v) {
        // documented 16-bit 16x32 A layout (lanes split by K-halves of 8)
        int kb = ((v < 4) ? (2 * v) : (16 + 2 * (v - 4))) + half * 8;
        fa[mi].u[v] = *(const uint32_t*)&sA[b][m][kb];
      }
    }
    #pragma unroll
    for (int g = 0; g < 4; ++g) {
      FragAB fb;
      const int n = wn * 16 + l16;
      #pragma unroll
      for (int v = 0; v < 8; ++v) {
        int k = ((v < 4) ? (2 * v) : (16 + 2 * (v - 4))) + half * 8;  // B 32x16: same K-halves-of-8 layout as A
        fb.u[v] = *(const uint32_t*)&sW[b][g][n][k];
      }
      #pragma unroll
      for (int mi = 0; mi < 2; ++mi)
        acc[g][mi] = __builtin_amdgcn_wmma_f32_16x16x32_bf16(
            false, fa[mi].v, false, fb.v, (short)0, acc[g][mi], false, false);
    }
  };

  // ---- prologue: stage chunk 0
  {
    uint32_t rw[8];
    asyncA(0, 0);
    loadW(0, rw);
    storeW(0, rw);
    asm volatile("s_wait_asynccnt 0x0" ::: "memory");
    __syncthreads();
  }

  // ---- pipelined main loop over 33 K-chunks (32 of h, 1 of x)
  for (int kt = 0; kt < 33; ++kt) {
    const int cur = kt & 1, nxt = (kt + 1) & 1;
    const bool haveNext = (kt < 32);
    const bool nextIsX  = (kt + 1 == 32);
    uint32_t rw[8], rax[8];

    if (haveNext) {
      if (!nextIsX) asyncA(kt + 1, nxt);
      else          loadAx(rax);
      loadW(kt + 1, rw);
      if (kt + 2 < 32)                        // gfx1250 global_prefetch_b8
        __builtin_prefetch(Whh + (size_t)(kt + 2) * 32 * NG + colBase, 0, 1);
    }

    compute(cur);

    if (haveNext) {
      storeW(nxt, rw);
      if (nextIsX) storeAx(nxt, rax);
    }
    asm volatile("s_wait_asynccnt 0x0" ::: "memory");
    __syncthreads();
  }

  // ---- epilogue: gates -> nonlinearities -> c/h update
  // C/D layout: lanes 0-15: N=lane, M=r ; lanes 16-31: N=lane-16, M=r+8
  const int gcol = colBase + wn * 16 + l16;
  const float bi = bias[0 * NH + gcol];
  const float bf = bias[1 * NH + gcol];
  const float bg = bias[2 * NH + gcol];
  const float bo = bias[3 * NH + gcol];

  #pragma unroll
  for (int mi = 0; mi < 2; ++mi) {
    #pragma unroll
    for (int r = 0; r < 8; ++r) {
      const int grow = rowBase + wm * 32 + mi * 16 + half * 8 + r;
      const size_t idx = (size_t)grow * NH + gcol;
      float iv = sigmoid_f(acc[0][mi][r] + bi);
      float fv = sigmoid_f(acc[1][mi][r] + bf);
      float gv = tanh_f(acc[2][mi][r] + bg);
      float ov = sigmoid_f(acc[3][mi][r] + bo);
      float cn = fv * cbuf[idx] + iv * gv;
      cbuf[idx] = cn;
      h_out[idx] = f32_to_bf16(ov * tanh_f(cn));
    }
  }
}

// ---------- tiny classifier: out = h @ Wc + bc ----------
__global__ __launch_bounds__(256) void classifier_kernel(
    const unsigned short* __restrict__ h,   // [NB][NH] bf16
    const float* __restrict__ Wc,           // [NH][NO]
    const float* __restrict__ bc,           // [NO]
    float* __restrict__ out)                // [NB][NO]
{
  int gidx = blockIdx.x * 256 + threadIdx.x;
  if (gidx >= NB * NO) return;
  int row = gidx / NO;
  int o   = gidx % NO;
  const unsigned short* hr = h + (size_t)row * NH;
  float s = bc[o];
  for (int k = 0; k < NH; ++k)
    s = fmaf(bf16_to_f32(hr[k]), Wc[k * NO + o], s);
  out[gidx] = s;
}

extern "C" void kernel_launch(void* const* d_in, const int* in_sizes, int n_in,
                              void* d_out, int out_size, void* d_ws, size_t ws_size,
                              hipStream_t stream) {
  const float* x    = (const float*)d_in[0];   // [B,T,IN]
  const float* W_ih = (const float*)d_in[1];   // [IN,4H]
  const float* W_hh = (const float*)d_in[2];   // [H,4H]
  const float* b    = (const float*)d_in[3];   // [4H]
  const float* Wc   = (const float*)d_in[4];   // [H,OUT]
  const float* bc   = (const float*)d_in[5];   // [OUT]

  // workspace layout (~48.5 MB total)
  char* ws = (char*)d_ws;
  unsigned short* whh_b = (unsigned short*)ws; ws += (size_t)NH * NG * 2;       // 8 MB
  unsigned short* wih_b = (unsigned short*)ws; ws += (size_t)NI * NG * 2;       // 224 KB
  unsigned short* xb    = (unsigned short*)ws; ws += (size_t)NT * NB * NI * 2;  // 6.1 MB
  unsigned short* hbuf0 = (unsigned short*)ws; ws += (size_t)NB * NH * 2;       // 8 MB
  unsigned short* hbuf1 = (unsigned short*)ws; ws += (size_t)NB * NH * 2;       // 8 MB
  float*          cbuf  = (float*)ws;          ws += (size_t)NB * NH * 4;       // 16 MB

  int n1 = NH * NG;
  cvt_f32_to_bf16_kernel<<<(n1 + 255) / 256, 256, 0, stream>>>(W_hh, whh_b, n1);
  int n2 = NI * NG;
  cvt_f32_to_bf16_kernel<<<(n2 + 255) / 256, 256, 0, stream>>>(W_ih, wih_b, n2);
  int n3 = NT * NB * NI;
  pack_x_kernel<<<(n3 + 255) / 256, 256, 0, stream>>>(x, xb);
  hipMemsetAsync(hbuf0, 0, (size_t)NB * NH * 2, stream);
  hipMemsetAsync(cbuf,  0, (size_t)NB * NH * 4, stream);

  unsigned short* hb[2] = {hbuf0, hbuf1};
  dim3 grid(NB / 128, NH / 32);   // 32 x 32 = 1024 blocks per step
  for (int t = 0; t < NT; ++t) {
    const unsigned short* hp = hb[t & 1];
    unsigned short*       ho = hb[(t + 1) & 1];
    lstm_step_kernel<<<grid, 256, 0, stream>>>(
        hp, xb + (size_t)t * NB * NI, whh_b, wih_b, b, cbuf, ho);
  }
  classifier_kernel<<<(NB * NO + 255) / 256, 256, 0, stream>>>(
      hb[NT & 1], Wc, bc, (float*)d_out);

  (void)in_sizes; (void)n_in; (void)out_size; (void)ws_size;
}
